// MultiHeadAttention_4638564679977
// MI455X (gfx1250) — matrix operand order repaired and checked by reading, not compiled, not run on hardware
//
#include <hip/hip_runtime.h>
#include <cstdint>
#include <cstddef>

#define D_MODEL 2048
#define N_HEADS 16
#define D_HEAD  128
#define SEQ     2048
#define BATCH   2
#define MTOT    (BATCH * SEQ)          // 4096 rows of activations
#define MBLK    32                     // GEMM rows per workgroup (TDM-staged in LDS)

typedef __attribute__((ext_vector_type(16))) __bf16        v16bf;
typedef __attribute__((ext_vector_type(8)))  float         v8f;
typedef __attribute__((ext_vector_type(8)))  unsigned int  v8u;
typedef __attribute__((ext_vector_type(4)))  unsigned int  u32x4;
typedef __attribute__((ext_vector_type(4)))  int           i32x4;
typedef __attribute__((ext_vector_type(8)))  int           i32x8;
typedef unsigned short u16;

#if defined(__has_builtin)
#if __has_builtin(__builtin_amdgcn_tensor_load_to_lds)
#define HAVE_TDM 1
#endif
#endif

// ---------- bf16 scalar helpers (round-to-nearest-even) ----------
__device__ __forceinline__ u16 f2bf(float f) {
  unsigned u = __builtin_bit_cast(unsigned, f);
  unsigned r = u + 0x7fffu + ((u >> 16) & 1u);
  return (u16)(r >> 16);
}
__device__ __forceinline__ float bf2f(u16 h) {
  return __builtin_bit_cast(float, (unsigned)h << 16);
}

// ---------- WMMA fragment loaders (bf16, 16x16x32) ----------
// A fragment (16x32, M x K). Caller passes: rowptr = A + (row0 + lane%16)*lda + k0 + 8*(lane/16)
// ISA layout: VGPR 0..3 hold K = {0..7}+8*half, VGPR 4..7 hold K = {16..23}+8*half.
__device__ __forceinline__ v16bf load_a16(const u16* base) {
  v8u u;
#pragma unroll
  for (int i = 0; i < 4; ++i) u[i]     = *(const unsigned int*)(base + 2 * i);
#pragma unroll
  for (int i = 0; i < 4; ++i) u[4 + i] = *(const unsigned int*)(base + 16 + 2 * i);
  return __builtin_bit_cast(v16bf, u);
}
// B fragment (32x16, K x N) sourced from a row-major (N, K) tensor (K contiguous).
// Caller passes: base = W + (col0 + lane%16)*ldw + k0 + 8*(lane/16)
__device__ __forceinline__ v16bf load_b16(const u16* base) {
  v8u u;
#pragma unroll
  for (int i = 0; i < 4; ++i) u[i]     = *(const unsigned int*)(base + 2 * i);
#pragma unroll
  for (int i = 0; i < 4; ++i) u[4 + i] = *(const unsigned int*)(base + 16 + 2 * i);
  return __builtin_bit_cast(v16bf, u);
}

__device__ __forceinline__ v8f wmma_bf16(v16bf a, v16bf b, v8f c) {
  return __builtin_amdgcn_wmma_f32_16x16x32_bf16(
      /*neg_a=*/false, a, /*neg_b=*/false, b,
      /*c_mod=*/(short)0, c, /*reuse_a=*/false, /*reuse_b=*/false);
}

// ---------- fp32 -> bf16 conversion ----------
__global__ __launch_bounds__(256) void f2bf_kernel(const float* __restrict__ src,
                                                   u16* __restrict__ dst, int n) {
  int i = blockIdx.x * 256 + threadIdx.x;
  if (i < n) dst[i] = f2bf(src[i]);
}

// ---------- GEMM: out(M=4096, N=2048) = A(4096x2048) * W^T + bias ----------
// A, W bf16; W row-major (N, K). Workgroup stages a 32-row A strip into LDS via the
// Tensor Data Mover, then each of 8 waves computes a 32x64 tile (8 wmma / k-step).
template <int STORE_BF16>
__global__ __launch_bounds__(256) void gemm_kernel(const u16* __restrict__ A,
                                                   const u16* __restrict__ W,
                                                   const float* __restrict__ bias,
                                                   void* __restrict__ out) {
  __shared__ u16 sA[MBLK * D_MODEL];  // 32 x 2048 bf16 = 128 KB (of 320 KB/WGP)

  const int wave = threadIdx.x >> 5;
  const int lane = threadIdx.x & 31;
  const int r    = lane & 15;
  const int half = lane >> 4;
  const int m0   = blockIdx.y * MBLK;
  const int n0   = (blockIdx.x * 8 + wave) * 64;

#if HAVE_TDM
  if (wave == 0) {
    // ---- Tensor DMA: A[m0 .. m0+31][0 .. 2047] -> LDS (2D D# descriptor) ----
    unsigned long long ga = (unsigned long long)(A + (size_t)m0 * D_MODEL);
    u32x4 g0;
    g0[0] = 1u;                                   // count=1, user mode, no gather
    g0[1] = 0u;                                   // lds_addr = 0 (sA at LDS base)
    g0[2] = (unsigned)(ga & 0xffffffffull);       // global_addr[31:0]
    g0[3] = (unsigned)((ga >> 32) & 0x01ffffffull) | (2u << 30);  // addr[56:32] | type=2
    i32x8 g1;
    g1[0] = (int)(1u << 16);                      // wg_mask=0, data_size=1 (2 bytes)
    g1[1] = (int)((D_MODEL & 0xffffu) << 16);     // tensor_dim0[15:0] @ bits 63:48
    g1[2] = (int)((D_MODEL >> 16) | ((MBLK & 0xffffu) << 16));  // dim0 hi | tensor_dim1 lo
    g1[3] = (int)((MBLK >> 16) | ((unsigned)D_MODEL << 16));    // dim1 hi | tile_dim0
    g1[4] = (int)MBLK;                            // tile_dim1=32, tile_dim2=0
    g1[5] = (int)D_MODEL;                         // tensor_dim0_stride[31:0]
    g1[6] = 0;                                    // stride0 hi | tensor_dim1_stride lo
    g1[7] = 0;
    i32x4 gz4 = {0, 0, 0, 0};                     // groups 2/3: dims unused (2D)
    i32x8 gz8 = {0, 0, 0, 0, 0, 0, 0, 0};
    __builtin_amdgcn_tensor_load_to_lds(g0, g1, gz4, gz4, gz8, 0);
    __builtin_amdgcn_s_wait_tensorcnt(0);
  }
#else
  // fallback: cooperative global -> LDS copy of the A strip
  for (int i = threadIdx.x; i < MBLK * D_MODEL / 2; i += 256) {
    ((unsigned*)sA)[i] = ((const unsigned*)(A + (size_t)m0 * D_MODEL))[i];
  }
#endif
  __syncthreads();

  v8f acc0[4], acc1[4];
#pragma unroll
  for (int j = 0; j < 4; ++j) {
    acc0[j] = (v8f){0, 0, 0, 0, 0, 0, 0, 0};
    acc1[j] = (v8f){0, 0, 0, 0, 0, 0, 0, 0};
  }

  const u16* sArow0 = sA + (size_t)r * D_MODEL + 8 * half;
  const u16* sArow1 = sA + (size_t)(16 + r) * D_MODEL + 8 * half;

  for (int k0 = 0; k0 < D_MODEL; k0 += 32) {
    v16bf a0 = load_a16(sArow0 + k0);
    v16bf a1 = load_a16(sArow1 + k0);
#pragma unroll
    for (int j = 0; j < 4; ++j) {
      const u16* wb = W + (size_t)(n0 + j * 16 + r) * D_MODEL + k0 + 8 * half;
      __builtin_prefetch(wb + 64, 0, 0);  // global_prefetch_b8, next-k W line
      v16bf b = load_b16(wb);
      acc0[j] = wmma_bf16(a0, b, acc0[j]);
      acc1[j] = wmma_bf16(a1, b, acc1[j]);
    }
  }

#pragma unroll
  for (int j = 0; j < 4; ++j) {
    float bv = bias[n0 + j * 16 + r];
#pragma unroll
    for (int v = 0; v < 8; ++v) {
      size_t i0 = (size_t)(m0 + v + 8 * half) * D_MODEL + n0 + j * 16 + r;
      size_t i1 = (size_t)(m0 + 16 + v + 8 * half) * D_MODEL + n0 + j * 16 + r;
      float x0 = acc0[j][v] + bv;
      float x1 = acc1[j][v] + bv;
      if (STORE_BF16) {
        ((u16*)out)[i0] = f2bf(x0);
        ((u16*)out)[i1] = f2bf(x1);
      } else {
        ((float*)out)[i0] = x0;
        ((float*)out)[i1] = x1;
      }
    }
  }
}

// ---------- RoPE + layout shuffle ----------
__global__ __launch_bounds__(256) void rope_kernel(const u16* __restrict__ Qr,
                                                   const u16* __restrict__ Kr,
                                                   const u16* __restrict__ Vr,
                                                   const float* __restrict__ sinT,
                                                   const float* __restrict__ cosT,
                                                   u16* __restrict__ Q,
                                                   u16* __restrict__ K,
                                                   u16* __restrict__ Vt) {
  int idx = blockIdx.x * 256 + threadIdx.x;  // B*H*T*64 threads
  int d = idx & 63;
  int t = (idx >> 6) & (SEQ - 1);
  int h = (idx >> 17) & (N_HEADS - 1);
  int b = idx >> 21;

  size_t src = (size_t)(b * SEQ + t) * D_MODEL + h * D_HEAD + d;
  size_t dst = (size_t)((b * N_HEADS + h) * SEQ + t) * D_HEAD + d;
  float c = cosT[t * 64 + d];
  float s = sinT[t * 64 + d];

  const float qs = 0.08838834764831845f;  // 1/sqrt(128), folded into Q
  float q1 = bf2f(Qr[src]), q2 = bf2f(Qr[src + 64]);
  Q[dst]      = f2bf((q1 * c - q2 * s) * qs);
  Q[dst + 64] = f2bf((q1 * s + q2 * c) * qs);

  float k1 = bf2f(Kr[src]), k2 = bf2f(Kr[src + 64]);
  K[dst]      = f2bf(k1 * c - k2 * s);
  K[dst + 64] = f2bf(k1 * s + k2 * c);

  size_t vdst = (size_t)((b * N_HEADS + h) * D_HEAD + d) * SEQ + t;
  Vt[vdst]                    = Vr[src];
  Vt[vdst + (size_t)64 * SEQ] = Vr[src + 64];
}

// ---------- causal flash attention ----------
__global__ __launch_bounds__(256) void attn_kernel(const u16* __restrict__ Q,
                                                   const u16* __restrict__ K,
                                                   const u16* __restrict__ Vt,
                                                   u16* __restrict__ O) {
  __shared__ u16 ptile[8 * 16 * 32];  // 1KB per wave: P-tile D->A transpose staging
  const int wave = threadIdx.x >> 5;
  const int lane = threadIdx.x & 31;
  const int wid  = blockIdx.x * 8 + wave;   // 4096 waves total
  const int qt   = wid & (SEQ / 16 - 1);
  const int h    = (wid >> 7) & (N_HEADS - 1);
  const int b    = wid >> 11;
  const int q0   = qt * 16;
  const int r    = lane & 15;
  const int half = lane >> 4;

  const u16* Qbh = Q  + (size_t)((b * N_HEADS + h) * SEQ) * D_HEAD;
  const u16* Kbh = K  + (size_t)((b * N_HEADS + h) * SEQ) * D_HEAD;
  const u16* Vbh = Vt + (size_t)((b * N_HEADS + h) * D_HEAD) * SEQ;

  v16bf qa[4];
#pragma unroll
  for (int ds = 0; ds < 4; ++ds)
    qa[ds] = load_a16(Qbh + (size_t)(q0 + r) * D_HEAD + ds * 32 + 8 * half);

  float m[8], l[8];
  v8f o[8];
#pragma unroll
  for (int v = 0; v < 8; ++v) { m[v] = -3.0e38f; l[v] = 0.f; }
#pragma unroll
  for (int j = 0; j < 8; ++j) o[j] = (v8f){0, 0, 0, 0, 0, 0, 0, 0};

  u16* pt = ptile + wave * (16 * 32);

  for (int k0 = 0; k0 <= q0 + 15; k0 += 32) {
    // ---- S = Q * K^T  (two 16-col tiles, 4 d-steps) ----
    v8f s0 = (v8f){0, 0, 0, 0, 0, 0, 0, 0};
    v8f s1 = (v8f){0, 0, 0, 0, 0, 0, 0, 0};
#pragma unroll
    for (int ds = 0; ds < 4; ++ds) {
      v16bf kb0 = load_b16(Kbh + (size_t)(k0 + r) * D_HEAD + ds * 32 + 8 * half);
      v16bf kb1 = load_b16(Kbh + (size_t)(k0 + 16 + r) * D_HEAD + ds * 32 + 8 * half);
      s0 = wmma_bf16(qa[ds], kb0, s0);
      s1 = wmma_bf16(qa[ds], kb1, s1);
    }

    // ---- causal mask + online softmax (row = v + 8*half, col = lane%16) ----
#pragma unroll
    for (int v = 0; v < 8; ++v) {
      const int row = q0 + v + 8 * half;
      float e0 = s0[v]; if (k0 + r > row)      e0 = -3.0e38f;
      float e1 = s1[v]; if (k0 + 16 + r > row) e1 = -3.0e38f;
      float mv = fmaxf(e0, e1);
      mv = fmaxf(mv, __shfl_xor(mv, 1, 32));
      mv = fmaxf(mv, __shfl_xor(mv, 2, 32));
      mv = fmaxf(mv, __shfl_xor(mv, 4, 32));
      mv = fmaxf(mv, __shfl_xor(mv, 8, 32));
      float mn = fmaxf(m[v], mv);
      float sc = __expf(m[v] - mn);
      float p0 = __expf(e0 - mn);
      float p1 = __expf(e1 - mn);
      float rs = p0 + p1;
      rs += __shfl_xor(rs, 1, 32);
      rs += __shfl_xor(rs, 2, 32);
      rs += __shfl_xor(rs, 4, 32);
      rs += __shfl_xor(rs, 8, 32);
      l[v] = l[v] * sc + rs;
      m[v] = mn;
#pragma unroll
      for (int j = 0; j < 8; ++j) o[j][v] *= sc;
      pt[(v + 8 * half) * 32 + r]      = f2bf(p0);
      pt[(v + 8 * half) * 32 + 16 + r] = f2bf(p1);
    }

    // same-wave DS ops are in-order: reload P as a 16x32 A fragment
    v16bf pa = load_a16(pt + r * 32 + 8 * half);

    // ---- O += P * V  (8 d-tiles of 16) ----
#pragma unroll
    for (int j = 0; j < 8; ++j) {
      v16bf vb = load_b16(Vbh + (size_t)(j * 16 + r) * SEQ + k0 + 8 * half);
      o[j] = wmma_bf16(pa, vb, o[j]);
    }
  }

  // ---- normalize and store O as bf16 (B, T, H*D) ----
#pragma unroll
  for (int v = 0; v < 8; ++v) {
    const int row = q0 + v + 8 * half;
    float inv = 1.0f / l[v];
#pragma unroll
    for (int j = 0; j < 8; ++j) {
      O[(size_t)(b * SEQ + row) * D_MODEL + h * D_HEAD + j * 16 + r] =
          f2bf(o[j][v] * inv);
    }
  }
}

// ------------------------------------------------------------------
extern "C" void kernel_launch(void* const* d_in, const int* in_sizes, int n_in,
                              void* d_out, int out_size, void* d_ws, size_t ws_size,
                              hipStream_t stream) {
  const float* x    = (const float*)d_in[0];
  const float* Wq   = (const float*)d_in[1];
  const float* bq   = (const float*)d_in[2];
  const float* Wk   = (const float*)d_in[3];
  const float* bk   = (const float*)d_in[4];
  const float* Wv   = (const float*)d_in[5];
  const float* bv   = (const float*)d_in[6];
  const float* Wo   = (const float*)d_in[7];
  const float* bo   = (const float*)d_in[8];
  const float* sinT = (const float*)d_in[9];
  const float* cosT = (const float*)d_in[10];

  const size_t nX = (size_t)MTOT * D_MODEL;     // 8,388,608
  const size_t nW = (size_t)D_MODEL * D_MODEL;  // 4,194,304

  char* ws = (char*)d_ws;
  size_t off = 0;
  auto take = [&](size_t bytes) -> void* {
    void* p = ws + off;
    off += (bytes + 255) & ~(size_t)255;
    return p;
  };
  u16* xb  = (u16*)take(nX * 2);
  u16* Wqb = (u16*)take(nW * 2);
  u16* Wkb = (u16*)take(nW * 2);
  u16* Wvb = (u16*)take(nW * 2);
  u16* Wob = (u16*)take(nW * 2);
  u16* Qr  = (u16*)take(nX * 2);
  u16* Kr  = (u16*)take(nX * 2);
  u16* Vr  = (u16*)take(nX * 2);
  u16* Qs  = (u16*)take(nX * 2);
  u16* Ks  = (u16*)take(nX * 2);
  u16* Vt  = (u16*)take(nX * 2);
  u16* Ob  = (u16*)take(nX * 2);

  // 1) fp32 -> bf16
  f2bf_kernel<<<(int)((nX + 255) / 256), 256, 0, stream>>>(x,  xb,  (int)nX);
  f2bf_kernel<<<(int)((nW + 255) / 256), 256, 0, stream>>>(Wq, Wqb, (int)nW);
  f2bf_kernel<<<(int)((nW + 255) / 256), 256, 0, stream>>>(Wk, Wkb, (int)nW);
  f2bf_kernel<<<(int)((nW + 255) / 256), 256, 0, stream>>>(Wv, Wvb, (int)nW);
  f2bf_kernel<<<(int)((nW + 255) / 256), 256, 0, stream>>>(Wo, Wob, (int)nW);

  // 2) QKV projections (x @ W^T + b), bf16 out; TDM-staged A strips
  dim3 gg(D_MODEL / (8 * 64), MTOT / MBLK);  // (4, 128)
  gemm_kernel<1><<<gg, 256, 0, stream>>>(xb, Wqb, bq, Qr);
  gemm_kernel<1><<<gg, 256, 0, stream>>>(xb, Wkb, bk, Kr);
  gemm_kernel<1><<<gg, 256, 0, stream>>>(xb, Wvb, bv, Vr);

  // 3) RoPE + per-head layouts (Q scaled by 1/sqrt(D), V transposed)
  rope_kernel<<<(BATCH * N_HEADS * SEQ * 64) / 256, 256, 0, stream>>>(
      Qr, Kr, Vr, sinT, cosT, Qs, Ks, Vt);

  // 4) causal flash attention -> Ob (B, T, H*D) bf16
  attn_kernel<<<(BATCH * N_HEADS * (SEQ / 16)) / 8, 256, 0, stream>>>(Qs, Ks, Vt, Ob);

  // 5) output projection, fp32 out
  gemm_kernel<0><<<gg, 256, 0, stream>>>(Ob, Wob, bo, (float*)d_out);
}
